// SakeInteractionBlock_9603546874393
// MI455X (gfx1250) — matrix operand order repaired and checked by reading, not compiled, not run on hardware
//
#include <hip/hip_runtime.h>
#include <hip/hip_bf16.h>

typedef __attribute__((ext_vector_type(16))) _Float16 v16h;
typedef __attribute__((ext_vector_type(8)))  float    v8f;

#define WMMA_F16(c, a, b) \
  c = __builtin_amdgcn_wmma_f32_16x16x32_f16(false, a, false, b, (short)0, c, false, false)

// ---------------- helpers ----------------
__device__ __forceinline__ float silu_f(float x) { return x / (1.0f + __expf(-x)); }
__device__ __forceinline__ float celu2_f(float x) {
  return fmaxf(x, 0.0f) + fminf(0.0f, 2.0f * (__expf(0.5f * x) - 1.0f));
}
// order-preserving float<->uint for atomicMax on floats (incl. negatives)
__device__ __forceinline__ unsigned enc_f(float f) {
  unsigned u = __float_as_uint(f);
  return (u & 0x80000000u) ? ~u : (u | 0x80000000u);
}
__device__ __forceinline__ float dec_f(unsigned e) {
  return (e & 0x80000000u) ? __uint_as_float(e & 0x7FFFFFFFu) : __uint_as_float(~e);
}

// A fragment (16x32 f16) from row-major LDS tile, leading dim ld, K offset k0.
// lane l: m = l%16; halves 0..7 -> K = k0 + 8*(l/16) + j ; halves 8..15 -> K = k0 + 16 + 8*(l/16) + j
__device__ __forceinline__ v16h frag_a_lds(const _Float16* t, int ld, int k0, int lane) {
  int m = lane & 15, g = (lane >> 4) & 1;
  const _Float16* row = t + m * ld + k0;
  v16h a;
#pragma unroll
  for (int j = 0; j < 8; ++j) a[j] = row[8 * g + j];
#pragma unroll
  for (int j = 0; j < 8; ++j) a[j + 8] = row[16 + 8 * g + j];
  return a;
}

// B fragment (32x16, B[k][n] = W[n][k]) from row-major weight W [O][Kp] (global f16).
// lane l: n = n0 + l%16; halves 0..7 -> K = k0 + 8*(l/16) + j ; halves 8..15 -> K = k0 + 16 + 8*(l/16) + j
__device__ __forceinline__ v16h frag_b_w(const _Float16* w, int ldk, int n0, int k0, int lane) {
  int n = n0 + (lane & 15), g = (lane >> 4) & 1;
  const _Float16* row = w + (size_t)n * ldk + k0 + 8 * g;
  v16h b;
#pragma unroll
  for (int j = 0; j < 16; ++j) b[j] = row[j + (j >> 3) * 8];
  return b;
}

// ---------------- constants ----------------
#define NATOMS 10000
#define NPAIRS 160000
#define HDIM 64
#define NHEAD 4
#define NRBF 50
#define NC 256

// ---------------- prep kernels ----------------
__global__ void convert_pad_f16(_Float16* dst, const float* src, int Od, int Os, int Kd, int Ks) {
  int t = blockIdx.x * blockDim.x + threadIdx.x;
  if (t >= Od * Kd) return;
  int o = t / Kd, k = t - o * Kd;
  float v = (o < Os && k < Ks) ? src[(size_t)o * Ks + k] : 0.0f;
  dst[t] = (_Float16)v;
}

__global__ void zero_f32(float* p, int n) {
  int t = blockIdx.x * blockDim.x + threadIdx.x;
  if (t < n) p[t] = 0.0f;
}

// ---------------- kernel 1: edge model (WMMA) ----------------
__global__ __launch_bounds__(32) void edge_kernel(
    const float* __restrict__ q, const float* __restrict__ d_ij,
    const int* __restrict__ idx_i, const int* __restrict__ idx_j,
    const float* __restrict__ rbf_off, const float* __restrict__ rbf_w,
    const _Float16* __restrict__ wf_in, const float* __restrict__ b_in,
    const _Float16* __restrict__ wf_out1, const float* __restrict__ b_out1,
    const _Float16* __restrict__ wf_out2, const float* __restrict__ b_out2,
    const float* __restrict__ w_att, const float* __restrict__ b_att,
    _Float16* __restrict__ qm16, float* __restrict__ att_out,
    unsigned* __restrict__ segmax) {
  const int p0 = blockIdx.x * 16;
  const int lane = threadIdx.x;
  __shared__ _Float16 sH[16 * 192];  // [q0(128) | filtered(50) | d(1) | pad(13)]
  __shared__ _Float16 sT[16 * 64];
  __shared__ float sQM[16 * 64];
  __shared__ float sD[16];
  __shared__ int sIJ[16];

  // gather q0 into sH cols 0..127
  for (int t = lane; t < 16 * 128; t += 32) {
    int r = t >> 7, c = t & 127;
    int atom = (c < 64) ? idx_i[p0 + r] : idx_j[p0 + r];
    sH[r * 192 + c] = (_Float16)q[(size_t)atom * HDIM + (c & 63)];
  }
  if (lane < 16) {
    float d = d_ij[p0 + lane];
    sD[lane] = d;
    sIJ[lane] = idx_j[p0 + lane];
    sH[lane * 192 + 178] = (_Float16)d;
  }
  for (int t = lane; t < 16 * 13; t += 32) {  // zero pad cols 179..191
    int r = t / 13, c = 179 + (t - r * 13);
    sH[r * 192 + c] = (_Float16)0.f;
  }
  __syncthreads();

  const int nl = lane & 15, g = (lane >> 4) & 1;

  // GEMM1: qh = q0 @ w_in^T  (K=128, N padded to 64), fuse RBF filter
  v16h a0 = frag_a_lds(sH, 192, 0, lane);
  v16h a1 = frag_a_lds(sH, 192, 32, lane);
  v16h a2 = frag_a_lds(sH, 192, 64, lane);
  v16h a3 = frag_a_lds(sH, 192, 96, lane);
  for (int ct = 0; ct < 4; ++ct) {
    v8f acc = {};
    WMMA_F16(acc, a0, frag_b_w(wf_in, 128, ct * 16, 0, lane));
    WMMA_F16(acc, a1, frag_b_w(wf_in, 128, ct * 16, 32, lane));
    WMMA_F16(acc, a2, frag_b_w(wf_in, 128, ct * 16, 64, lane));
    WMMA_F16(acc, a3, frag_b_w(wf_in, 128, ct * 16, 96, lane));
    int n = ct * 16 + nl;
    if (n < NRBF) {
      float off = rbf_off[n], wd = rbf_w[n];
      float coeff = -0.5f / (wd * wd);
      float bb = b_in[n];
#pragma unroll
      for (int v = 0; v < 8; ++v) {
        int m = 8 * g + v;
        float dd = sD[m] - off;
        float rbf = __expf(coeff * dd * dd);
        sH[m * 192 + 128 + n] = (_Float16)(rbf * (acc[v] + bb));
      }
    }
  }
  __syncthreads();

  // GEMM2: t = silu(h @ w_out1^T + b)  (K=192 padded, N=64)
  v16h a4 = frag_a_lds(sH, 192, 128, lane);
  v16h a5 = frag_a_lds(sH, 192, 160, lane);
  for (int ct = 0; ct < 4; ++ct) {
    v8f acc = {};
    WMMA_F16(acc, a0, frag_b_w(wf_out1, 192, ct * 16, 0, lane));
    WMMA_F16(acc, a1, frag_b_w(wf_out1, 192, ct * 16, 32, lane));
    WMMA_F16(acc, a2, frag_b_w(wf_out1, 192, ct * 16, 64, lane));
    WMMA_F16(acc, a3, frag_b_w(wf_out1, 192, ct * 16, 96, lane));
    WMMA_F16(acc, a4, frag_b_w(wf_out1, 192, ct * 16, 128, lane));
    WMMA_F16(acc, a5, frag_b_w(wf_out1, 192, ct * 16, 160, lane));
    int n = ct * 16 + nl;
    float bb = b_out1[n];
#pragma unroll
    for (int v = 0; v < 8; ++v) sT[(8 * g + v) * 64 + n] = (_Float16)silu_f(acc[v] + bb);
  }
  __syncthreads();

  // GEMM3: q_ij_mtx = t @ w_out2^T + b  (K=64, N=64)
  v16h t0 = frag_a_lds(sT, 64, 0, lane);
  v16h t1 = frag_a_lds(sT, 64, 32, lane);
  for (int ct = 0; ct < 4; ++ct) {
    v8f acc = {};
    WMMA_F16(acc, t0, frag_b_w(wf_out2, 64, ct * 16, 0, lane));
    WMMA_F16(acc, t1, frag_b_w(wf_out2, 64, ct * 16, 32, lane));
    int n = ct * 16 + nl;
    float bb = b_out2[n];
#pragma unroll
    for (int v = 0; v < 8; ++v) {
      int m = 8 * g + v;
      float val = acc[v] + bb;
      sQM[m * 64 + n] = val;
      qm16[(size_t)(p0 + m) * 64 + n] = (_Float16)val;
    }
  }
  __syncthreads();

  // attention logits: att = celu(q_ij_mtx @ w_att^T + b, alpha=2); segment max
  for (int t = lane; t < 16 * NHEAD; t += 32) {
    int p = t >> 2, h = t & 3;
    float s = b_att[h];
    for (int k = 0; k < HDIM; ++k) s += sQM[p * 64 + k] * w_att[h * HDIM + k];
    float att = celu2_f(s);
    att_out[(size_t)(p0 + p) * NHEAD + h] = att;
    atomicMax(&segmax[sIJ[p] * NHEAD + h], enc_f(att));
  }
}

// ---------------- kernel 2: softmax denominators + counts ----------------
__global__ void softmax_pass2(const int* __restrict__ idx_j,
                              const unsigned* __restrict__ segmax,
                              float* __restrict__ att_e,  // in: logits, out: exp
                              float* __restrict__ denom, float* __restrict__ cnt) {
  int t = blockIdx.x * blockDim.x + threadIdx.x;
  if (t >= NPAIRS * NHEAD) return;
  int p = t >> 2, h = t & 3;
  int n = idx_j[p];
  float e = __expf(att_e[t] - dec_f(segmax[n * NHEAD + h]));
  att_e[t] = e;
  atomicAdd(&denom[n * NHEAD + h], e);
  if (h == 0) atomicAdd(&cnt[n], 1.0f);
}

// ---------------- kernel 3: q_ij_att, mix GEMM, spatial scatter ----------------
__global__ __launch_bounds__(32) void mix_kernel(
    const _Float16* __restrict__ qm16, const float* __restrict__ e_buf,
    const float* __restrict__ denom, const int* __restrict__ idx_j,
    const float* __restrict__ r_ij, const float* __restrict__ d_ij,
    const _Float16* __restrict__ wf_mix,
    float* __restrict__ qij, float* __restrict__ sums) {
  const int p0 = blockIdx.x * 16;
  const int lane = threadIdx.x;
  __shared__ _Float16 sQA[16 * 256];
  __shared__ float sSem[16 * NHEAD];
  __shared__ float sRH[16 * 3];
  __shared__ int sIJ[16];

  if (lane < 16) sIJ[lane] = idx_j[p0 + lane];
  for (int t = lane; t < 16 * NHEAD; t += 32) {
    int p = t >> 2, h = t & 3;
    sSem[t] = e_buf[(size_t)(p0 + p) * NHEAD + h] / denom[idx_j[p0 + p] * NHEAD + h];
  }
  for (int t = lane; t < 16 * 3; t += 32) {
    int p = t / 3, d = t - p * 3;
    sRH[t] = r_ij[(size_t)(p0 + p) * 3 + d] / (d_ij[p0 + p] + 1e-5f);
  }
  __syncthreads();

  // q_ij_att[p, f*4+h] = q_ij_mtx[p,f] * sem[p,h]; scatter-add into q_ij
  for (int t = lane; t < 16 * NC; t += 32) {
    int r = t >> 8, c = t & 255;
    float val = (float)qm16[(size_t)(p0 + r) * 64 + (c >> 2)] * sSem[r * 4 + (c & 3)];
    sQA[t] = (_Float16)val;
    atomicAdd(&qij[(size_t)sIJ[r] * NC + c], val);
  }
  __syncthreads();

  const int nl = lane & 15, g = (lane >> 4) & 1;
  v16h af[8];
#pragma unroll
  for (int ks = 0; ks < 8; ++ks) af[ks] = frag_a_lds(sQA, 256, ks * 32, lane);

  // coeffs = tanh(q_ij_att @ w_mix^T)  (16 x 256 x 256), fused scatter of coeffs x r_hat
  for (int ct = 0; ct < 16; ++ct) {
    v8f acc = {};
#pragma unroll
    for (int ks = 0; ks < 8; ++ks)
      WMMA_F16(acc, af[ks], frag_b_w(wf_mix, 256, ct * 16, ks * 32, lane));
    int c = ct * 16 + nl;
#pragma unroll
    for (int v = 0; v < 8; ++v) {
      int m = 8 * g + v;
      float co = tanhf(acc[v]);
      size_t base = (size_t)sIJ[m] * (NC * 3) + (size_t)c * 3;
      atomicAdd(&sums[base + 0], co * sRH[m * 3 + 0]);
      atomicAdd(&sums[base + 1], co * sRH[m * 3 + 1]);
      atomicAdd(&sums[base + 2], co * sRH[m * 3 + 2]);
    }
  }
}

// ---------------- kernel 4: node model (WMMA) ----------------
__global__ __launch_bounds__(32) void node_kernel(
    const float* __restrict__ q, const float* __restrict__ sums,
    const float* __restrict__ cnt, const float* __restrict__ qij,
    const _Float16* __restrict__ wf_post1, const float* __restrict__ b_post1,
    const _Float16* __restrict__ wf_post2, const float* __restrict__ b_post2,
    const _Float16* __restrict__ wf_node1, const float* __restrict__ b_node1,
    const _Float16* __restrict__ wf_node2, const float* __restrict__ b_node2,
    float* __restrict__ out) {
  const int a0 = blockIdx.x * 16;
  const int lane = threadIdx.x;
  __shared__ _Float16 sN[16 * 256];   // ||mean||^2
  __shared__ _Float16 sH2[16 * 384];  // [q | q_ij | q_comb]
  __shared__ _Float16 sT4[16 * 64];
  __shared__ float sInv[16];

  if (lane < 16) sInv[lane] = 1.0f / fmaxf(cnt[a0 + lane], 1.0f);
  __syncthreads();

  for (int t = lane; t < 16 * NC; t += 32) {
    int r = t >> 8, c = t & 255;
    size_t b = (size_t)(a0 + r) * (NC * 3) + (size_t)c * 3;
    float inv = sInv[r];
    float x = sums[b] * inv, y = sums[b + 1] * inv, z = sums[b + 2] * inv;
    sN[t] = (_Float16)(x * x + y * y + z * z);
    sH2[r * 384 + 64 + c] = (_Float16)qij[(size_t)(a0 + r) * NC + c];
  }
  for (int t = lane; t < 16 * 64; t += 32) {
    int r = t >> 6, c = t & 63;
    sH2[r * 384 + c] = (_Float16)q[(size_t)(a0 + r) * HDIM + c];
  }
  __syncthreads();

  const int nl = lane & 15, g = (lane >> 4) & 1;
  // post1: silu(norm @ w_post1^T + b)   (K=256, N=64)
  {
    v16h af[8];
#pragma unroll
    for (int ks = 0; ks < 8; ++ks) af[ks] = frag_a_lds(sN, 256, ks * 32, lane);
    for (int ct = 0; ct < 4; ++ct) {
      v8f acc = {};
#pragma unroll
      for (int ks = 0; ks < 8; ++ks)
        WMMA_F16(acc, af[ks], frag_b_w(wf_post1, 256, ct * 16, ks * 32, lane));
      int n = ct * 16 + nl;
      float bb = b_post1[n];
#pragma unroll
      for (int v = 0; v < 8; ++v) sT4[(8 * g + v) * 64 + n] = (_Float16)silu_f(acc[v] + bb);
    }
  }
  __syncthreads();
  // post2: q_comb -> sH2 cols 320..383
  {
    v16h t0 = frag_a_lds(sT4, 64, 0, lane);
    v16h t1 = frag_a_lds(sT4, 64, 32, lane);
    for (int ct = 0; ct < 4; ++ct) {
      v8f acc = {};
      WMMA_F16(acc, t0, frag_b_w(wf_post2, 64, ct * 16, 0, lane));
      WMMA_F16(acc, t1, frag_b_w(wf_post2, 64, ct * 16, 32, lane));
      int n = ct * 16 + nl;
      float bb = b_post2[n];
#pragma unroll
      for (int v = 0; v < 8; ++v)
        sH2[(8 * g + v) * 384 + 320 + n] = (_Float16)silu_f(acc[v] + bb);
    }
  }
  __syncthreads();
  // node1: silu(h2 @ w_node1^T + b)  (K=384, N=64)
  {
    v16h af[12];
#pragma unroll
    for (int ks = 0; ks < 12; ++ks) af[ks] = frag_a_lds(sH2, 384, ks * 32, lane);
    for (int ct = 0; ct < 4; ++ct) {
      v8f acc = {};
#pragma unroll
      for (int ks = 0; ks < 12; ++ks)
        WMMA_F16(acc, af[ks], frag_b_w(wf_node1, 384, ct * 16, ks * 32, lane));
      int n = ct * 16 + nl;
      float bb = b_node1[n];
#pragma unroll
      for (int v = 0; v < 8; ++v) sT4[(8 * g + v) * 64 + n] = (_Float16)silu_f(acc[v] + bb);
    }
  }
  __syncthreads();
  // node2 + residual
  {
    v16h t0 = frag_a_lds(sT4, 64, 0, lane);
    v16h t1 = frag_a_lds(sT4, 64, 32, lane);
    for (int ct = 0; ct < 4; ++ct) {
      v8f acc = {};
      WMMA_F16(acc, t0, frag_b_w(wf_node2, 64, ct * 16, 0, lane));
      WMMA_F16(acc, t1, frag_b_w(wf_node2, 64, ct * 16, 32, lane));
      int n = ct * 16 + nl;
      float bb = b_node2[n];
#pragma unroll
      for (int v = 0; v < 8; ++v) {
        size_t o = (size_t)(a0 + 8 * g + v) * HDIM + n;
        out[o] = q[o] + silu_f(acc[v] + bb);
      }
    }
  }
}

// ---------------- launch ----------------
extern "C" void kernel_launch(void* const* d_in, const int* in_sizes, int n_in,
                              void* d_out, int out_size, void* d_ws, size_t ws_size,
                              hipStream_t stream) {
  const float* q       = (const float*)d_in[0];
  // d_in[1] = mu (unused by reference)
  const float* r_ij    = (const float*)d_in[2];
  const float* d_ij    = (const float*)d_in[3];
  const int*   idx_i   = (const int*)d_in[4];
  const int*   idx_j   = (const int*)d_in[5];
  const float* rbf_off = (const float*)d_in[6];
  const float* rbf_w   = (const float*)d_in[7];
  const float* w_in    = (const float*)d_in[8];
  const float* b_in    = (const float*)d_in[9];
  const float* w_out1  = (const float*)d_in[10];
  const float* b_out1  = (const float*)d_in[11];
  const float* w_out2  = (const float*)d_in[12];
  const float* b_out2  = (const float*)d_in[13];
  const float* w_att   = (const float*)d_in[14];
  const float* b_att   = (const float*)d_in[15];
  const float* w_mix   = (const float*)d_in[16];
  const float* w_post1 = (const float*)d_in[17];
  const float* b_post1 = (const float*)d_in[18];
  const float* w_post2 = (const float*)d_in[19];
  const float* b_post2 = (const float*)d_in[20];
  const float* w_node1 = (const float*)d_in[21];
  const float* b_node1 = (const float*)d_in[22];
  const float* w_node2 = (const float*)d_in[23];
  const float* b_node2 = (const float*)d_in[24];

  char* ws = (char*)d_ws;
  size_t off = 0;
  auto alloc = [&](size_t bytes) -> void* {
    void* p = ws + off;
    off += (bytes + 255) & ~(size_t)255;
    return p;
  };
  _Float16* wf_in    = (_Float16*)alloc(64 * 128 * 2);
  _Float16* wf_out1  = (_Float16*)alloc(64 * 192 * 2);
  _Float16* wf_out2  = (_Float16*)alloc(64 * 64 * 2);
  _Float16* wf_mix   = (_Float16*)alloc(256 * 256 * 2);
  _Float16* wf_post1 = (_Float16*)alloc(64 * 256 * 2);
  _Float16* wf_post2 = (_Float16*)alloc(64 * 64 * 2);
  _Float16* wf_node1 = (_Float16*)alloc(64 * 384 * 2);
  _Float16* wf_node2 = (_Float16*)alloc(64 * 64 * 2);
  _Float16* qm16     = (_Float16*)alloc((size_t)NPAIRS * 64 * 2);
  float*    att_e    = (float*)alloc((size_t)NPAIRS * NHEAD * 4);
  unsigned* segmax   = (unsigned*)alloc((size_t)NATOMS * NHEAD * 4);
  float*    denom    = (float*)alloc((size_t)NATOMS * NHEAD * 4);
  float*    cnt      = (float*)alloc((size_t)NATOMS * 4);
  float*    qij      = (float*)alloc((size_t)NATOMS * NC * 4);
  float*    sums     = (float*)alloc((size_t)NATOMS * NC * 3 * 4);
  (void)ws_size; (void)in_sizes; (void)n_in; (void)out_size;

  // weight conversion (f16 + padding)
  auto cvt = [&](_Float16* dst, const float* src, int Od, int Os, int Kd, int Ks) {
    int n = Od * Kd;
    convert_pad_f16<<<(n + 255) / 256, 256, 0, stream>>>(dst, src, Od, Os, Kd, Ks);
  };
  cvt(wf_in, w_in, 64, 50, 128, 128);
  cvt(wf_out1, w_out1, 64, 64, 192, 179);
  cvt(wf_out2, w_out2, 64, 64, 64, 64);
  cvt(wf_mix, w_mix, 256, 256, 256, 256);
  cvt(wf_post1, w_post1, 64, 64, 256, 256);
  cvt(wf_post2, w_post2, 64, 64, 64, 64);
  cvt(wf_node1, w_node1, 64, 64, 384, 384);
  cvt(wf_node2, w_node2, 64, 64, 64, 64);

  // zero accumulators (bit pattern 0 valid for the encoded segmax as -inf sentinel)
  auto zro = [&](void* p, int n) { zero_f32<<<(n + 255) / 256, 256, 0, stream>>>((float*)p, n); };
  zro(segmax, NATOMS * NHEAD);
  zro(denom, NATOMS * NHEAD);
  zro(cnt, NATOMS);
  zro(qij, NATOMS * NC);
  zro(sums, NATOMS * NC * 3);

  edge_kernel<<<NPAIRS / 16, 32, 0, stream>>>(
      q, d_ij, idx_i, idx_j, rbf_off, rbf_w,
      wf_in, b_in, wf_out1, b_out1, wf_out2, b_out2,
      w_att, b_att, qm16, att_e, segmax);

  softmax_pass2<<<(NPAIRS * NHEAD + 255) / 256, 256, 0, stream>>>(
      idx_j, segmax, att_e, denom, cnt);

  mix_kernel<<<NPAIRS / 16, 32, 0, stream>>>(
      qm16, att_e, denom, idx_j, r_ij, d_ij, wf_mix, qij, sums);

  node_kernel<<<NATOMS / 16, 32, 0, stream>>>(
      q, sums, cnt, qij,
      wf_post1, b_post1, wf_post2, b_post2,
      wf_node1, b_node1, wf_node2, b_node2,
      (float*)d_out);
}
